// AttentionHead_80436147520097
// MI455X (gfx1250) — matrix operand order repaired and checked by reading, not compiled, not run on hardware
//
#include <hip/hip_runtime.h>
#include <math.h>

#define HEAD   64
#define DIN    1024
#define BB     4
#define TT     4096
#define MTOT   (BB * TT)

typedef __attribute__((ext_vector_type(16))) __bf16 v16bf;
typedef __attribute__((ext_vector_type(8)))  __bf16 v8bf;
typedef __attribute__((ext_vector_type(8)))  float  v8f;
typedef __attribute__((ext_vector_type(4)))  float  v4f;

union V16 { v16bf v; v8bf h[2]; };

static __device__ __forceinline__ v8f wmma_bf16(v16bf a, v16bf b, v8f c) {
    // D = A(16x32 bf16) * B(32x16 bf16) + C(16x16 f32)
    return __builtin_amdgcn_wmma_f32_16x16x32_bf16(
        /*neg_a=*/false, a, /*neg_b=*/false, b,
        /*c_mod=*/(short)0, c, /*reuse_a=*/false, /*reuse_b=*/false);
}

// Assemble a v16bf A/B fragment from two contiguous 16-byte chunks.
static __device__ __forceinline__ v16bf pack2(const __bf16* lo, const __bf16* hi) {
    V16 u;
    u.h[0] = *(const v8bf*)lo;
    u.h[1] = *(const v8bf*)hi;
    return u.v;
}

// ---------------------------------------------------------------------------
// Kernel 0: transpose + convert the three 1024x64 fp32 weights into
// bf16 Wt[3][64][1024] (column-of-W becomes contiguous K for the B operand).
// ---------------------------------------------------------------------------
__global__ __launch_bounds__(256) void prep_weights(
    const float* __restrict__ Wk, const float* __restrict__ Wq,
    const float* __restrict__ Wv, __bf16* __restrict__ Wt) {
    int idx = blockIdx.x * 256 + threadIdx.x;
    if (idx >= 3 * HEAD * DIN) return;
    int w   = idx / (HEAD * DIN);
    int rem = idx % (HEAD * DIN);
    int nn  = rem / DIN;          // output column (0..63)
    int kk  = rem % DIN;          // reduction index (0..1023)
    const float* W = (w == 0) ? Wk : (w == 1) ? Wq : Wv;
    Wt[idx] = (__bf16)W[(size_t)kk * HEAD + nn];
}

// A-operand fragment from a row-major fp32 row: slots 0..15 = K[k0+16h .. +16),
// converted to bf16 in-register.
static __device__ __forceinline__ v16bf load_a_x(const float* xrow, int k0, int h) {
    v4f f0 = *(const v4f*)(xrow + k0 + 16 * h);
    v4f f1 = *(const v4f*)(xrow + k0 + 16 * h + 4);
    v4f g0 = *(const v4f*)(xrow + k0 + 16 * h + 8);
    v4f g1 = *(const v4f*)(xrow + k0 + 16 * h + 12);
    v16bf a;
#pragma unroll
    for (int i = 0; i < 4; i++) {
        a[i]      = (__bf16)f0[i];
        a[4 + i]  = (__bf16)f1[i];
        a[8 + i]  = (__bf16)g0[i];
        a[12 + i] = (__bf16)g1[i];
    }
    return a;
}

// ---------------------------------------------------------------------------
// Kernel 1: projections. grid = (MTOT/128, 3). 256 threads = 8 waves.
// Wave computes a 16(M) x 64(N) tile over K=1024 in steps of 32, with a
// 2x-unrolled ping-pong pipeline (no register-rotation copies).
// ---------------------------------------------------------------------------
__global__ __launch_bounds__(256) void proj_kernel(
    const float* __restrict__ x, const __bf16* __restrict__ Wt,
    const float* __restrict__ bk, const float* __restrict__ bq,
    const float* __restrict__ bv,
    __bf16* __restrict__ kb, __bf16* __restrict__ qb,
    __bf16* __restrict__ vt) {
    const int wsel = blockIdx.y;              // 0=k, 1=q, 2=v
    const int wave = threadIdx.x >> 5;
    const int lane = threadIdx.x & 31;
    const int n    = lane & 15;
    const int h    = lane >> 4;
    const int m0   = blockIdx.x * 128 + wave * 16;   // tile row base

    const __bf16* W    = Wt + (size_t)wsel * HEAD * DIN;
    const float*  bias = (wsel == 0) ? bk : (wsel == 1) ? bq : bv;
    const float*  xrow = x + (size_t)(m0 + n) * DIN;  // A row = lane%16

    v8f acc[4];
#pragma unroll
    for (int nt = 0; nt < 4; nt++) acc[nt] = (v8f)0.0f;

    // prologue: operand set 0 holds k0 = 0
    v16bf a0 = load_a_x(xrow, 0, h);
    v16bf b0[4];
#pragma unroll
    for (int nt = 0; nt < 4; nt++) {
        const __bf16* wp = W + (size_t)(16 * nt + n) * DIN + 16 * h;
        b0[nt] = pack2(wp, wp + 8);
    }

    v16bf a1;
    v16bf b1[4];
#pragma unroll 1
    for (int k0 = 0; k0 < DIN - 64; k0 += 64) {
        // load set 1 (k0+32) while WMMAs consume set 0 (k0)
        a1 = load_a_x(xrow, k0 + 32, h);
#pragma unroll
        for (int nt = 0; nt < 4; nt++) {
            const __bf16* wp = W + (size_t)(16 * nt + n) * DIN + k0 + 32 + 16 * h;
            b1[nt] = pack2(wp, wp + 8);
        }
#pragma unroll
        for (int nt = 0; nt < 4; nt++)
            acc[nt] = wmma_bf16(a0, b0[nt], acc[nt]);

        // load set 0 (k0+64) while WMMAs consume set 1 (k0+32)
        a0 = load_a_x(xrow, k0 + 64, h);
#pragma unroll
        for (int nt = 0; nt < 4; nt++) {
            const __bf16* wp = W + (size_t)(16 * nt + n) * DIN + k0 + 64 + 16 * h;
            b0[nt] = pack2(wp, wp + 8);
        }
#pragma unroll
        for (int nt = 0; nt < 4; nt++)
            acc[nt] = wmma_bf16(a1, b1[nt], acc[nt]);
    }
    // tail: set 0 holds DIN-64; load DIN-32 into set 1, then drain both
    a1 = load_a_x(xrow, DIN - 32, h);
#pragma unroll
    for (int nt = 0; nt < 4; nt++) {
        const __bf16* wp = W + (size_t)(16 * nt + n) * DIN + DIN - 32 + 16 * h;
        b1[nt] = pack2(wp, wp + 8);
    }
#pragma unroll
    for (int nt = 0; nt < 4; nt++)
        acc[nt] = wmma_bf16(a0, b0[nt], acc[nt]);
#pragma unroll
    for (int nt = 0; nt < 4; nt++)
        acc[nt] = wmma_bf16(a1, b1[nt], acc[nt]);

    const int bidx  = m0 / TT;    // 128-row tiles never straddle T=4096
    const int tbase = m0 % TT;

#pragma unroll
    for (int nt = 0; nt < 4; nt++) {
        const int   col = 16 * nt + n;
        const float bv_ = bias[col];
        if (wsel == 2) {
            // store V transposed: vt[b][col][t]; r=0..7 are consecutive t
            v8bf pv;
#pragma unroll
            for (int r = 0; r < 8; r++) pv[r] = (__bf16)(acc[nt][r] + bv_);
            *(v8bf*)(vt + ((size_t)bidx * HEAD + col) * TT + tbase + 8 * h) = pv;
        } else {
            __bf16* dst = (wsel == 0) ? kb : qb;
#pragma unroll
            for (int r = 0; r < 8; r++)
                dst[(size_t)(m0 + r + 8 * h) * HEAD + col] = (__bf16)(acc[nt][r] + bv_);
        }
    }
}

// ---------------------------------------------------------------------------
// Kernel 2: fused causal attention, flash-style online softmax.
// grid = (T/128, B), 256 threads = 8 waves; each wave owns 16 query rows t
// and streams key blocks of 64 columns s (amortizes softmax VALU over 2x
// the WMMA work vs 32-wide blocks). S[t,s] = k_t . q_s (reference order).
// ---------------------------------------------------------------------------
#define PSTR 72   // LDS row stride in bf16: 36 dwords -> conflict-free b128 reads

__global__ __launch_bounds__(256) void attn_kernel(
    const __bf16* __restrict__ kb, const __bf16* __restrict__ qb,
    const __bf16* __restrict__ vt, float* __restrict__ out) {
    __shared__ __bf16 plds[8][16 * PSTR];    // per-wave P staging (18 KB total)

    const int wave = threadIdx.x >> 5;
    const int lane = threadIdx.x & 31;
    const int n    = lane & 15;
    const int h    = lane >> 4;
    const int b    = blockIdx.y;
    const int t0   = blockIdx.x * 128 + wave * 16;

    const __bf16* K  = kb + (size_t)b * TT * HEAD;
    const __bf16* Q  = qb + (size_t)b * TT * HEAD;
    const __bf16* Vt = vt + (size_t)b * HEAD * TT;
    __bf16* myP = plds[wave];

    // A operand: 16 K-rows (t0..t0+15), 64 features in two K=32 steps
    v16bf aK[2];
#pragma unroll
    for (int ks = 0; ks < 2; ks++) {
        const __bf16* kr = K + (size_t)(t0 + n) * HEAD + 32 * ks;
        aK[ks] = pack2(kr + 16 * h, kr + 16 * h + 8);
    }

    float mrun[8], lrun[8];
    v8f   o[4];
#pragma unroll
    for (int r = 0; r < 8; r++) { mrun[r] = -INFINITY; lrun[r] = 0.0f; }
#pragma unroll
    for (int nt = 0; nt < 4; nt++) o[nt] = (v8f)0.0f;

    const float scale = 0.125f;   // 1/sqrt(64)

    // s0 is a multiple of 64 and <= 4032, so all s reads stay < T.
    for (int s0 = 0; s0 <= t0 + 15; s0 += 64) {
        // speculative prefetch of the NEXT key block (Q rows + Vt rows)
        const int spf = s0 + 64;
        if (spf <= t0 + 15) {
            __builtin_prefetch(Q + (size_t)(spf + lane) * HEAD, 0, 1);
            __builtin_prefetch(Q + (size_t)(spf + 32 + lane) * HEAD, 0, 1);
            __builtin_prefetch(Vt + (size_t)lane * TT + spf, 0, 1);
            __builtin_prefetch(Vt + (size_t)(lane + 32) * TT + spf, 0, 1);
        }

        // ---- batch all Q operand loads, then the 8 score WMMAs
        v16bf bq[4][2];
#pragma unroll
        for (int c = 0; c < 4; c++)
#pragma unroll
            for (int ks = 0; ks < 2; ks++) {
                const __bf16* qp = Q + (size_t)(s0 + 16 * c + n) * HEAD + 32 * ks + 16 * h;
                bq[c][ks] = pack2(qp, qp + 8);
            }
        v8f S[4];
#pragma unroll
        for (int c = 0; c < 4; c++) S[c] = (v8f)0.0f;
#pragma unroll
        for (int c = 0; c < 4; c++)
#pragma unroll
            for (int ks = 0; ks < 2; ks++)
                S[c] = wmma_bf16(aK[ks], bq[c][ks], S[c]);

        // ---- scale + causal mask + row max over the 64 columns
        float rmax[8];
#pragma unroll
        for (int r = 0; r < 8; r++) rmax[r] = -INFINITY;
#pragma unroll
        for (int c = 0; c < 4; c++) {
            const bool need_mask = (s0 + 16 * c + 15 > t0);
#pragma unroll
            for (int r = 0; r < 8; r++) {
                float v = S[c][r] * scale;
                if (need_mask && (s0 + 16 * c + n > t0 + r + 8 * h)) v = -INFINITY;
                S[c][r] = v;
                rmax[r] = fmaxf(rmax[r], v);
            }
        }
#pragma unroll
        for (int r = 0; r < 8; r++)
#pragma unroll
            for (int off = 1; off < 16; off <<= 1)
                rmax[r] = fmaxf(rmax[r], __shfl_xor(rmax[r], off));

        // ---- online softmax update
        float alpha[8], rsum[8];
#pragma unroll
        for (int r = 0; r < 8; r++) {
            float mnew = fmaxf(mrun[r], rmax[r]);
            alpha[r]   = __expf(mrun[r] - mnew);
            mrun[r]    = mnew;
            rsum[r]    = 0.0f;
        }
#pragma unroll
        for (int c = 0; c < 4; c++)
#pragma unroll
            for (int r = 0; r < 8; r++) {
                float p = __expf(S[c][r] - mrun[r]);
                S[c][r] = p;
                rsum[r] += p;
            }
#pragma unroll
        for (int r = 0; r < 8; r++)
#pragma unroll
            for (int off = 1; off < 16; off <<= 1)
                rsum[r] += __shfl_xor(rsum[r], off);
#pragma unroll
        for (int r = 0; r < 8; r++) lrun[r] = lrun[r] * alpha[r] + rsum[r];
#pragma unroll
        for (int nt = 0; nt < 4; nt++)
#pragma unroll
            for (int r = 0; r < 8; r++) o[nt][r] *= alpha[r];

        // ---- P (C-layout) -> LDS row-major -> A-layout reload (per-wave,
        // same-wave DS ordering makes this barrier-free)
#pragma unroll
        for (int c = 0; c < 4; c++)
#pragma unroll
            for (int r = 0; r < 8; r++)
                myP[(r + 8 * h) * PSTR + 16 * c + n] = (__bf16)S[c][r];

        v16bf aP[2];
#pragma unroll
        for (int g = 0; g < 2; g++) {
            const __bf16* pr = myP + n * PSTR + 32 * g;
            aP[g] = pack2(pr + 16 * h, pr + 16 * h + 8);
        }

        // ---- O += P(16x64) * V(64x64); B operand from transposed V
#pragma unroll
        for (int nt = 0; nt < 4; nt++)
#pragma unroll
            for (int ks = 0; ks < 2; ks++) {
                const __bf16* vp = Vt + (size_t)(16 * nt + n) * TT + s0 + 32 * ks + 16 * h;
                v16bf bV = pack2(vp, vp + 8);
                o[nt] = wmma_bf16(aP[ks], bV, o[nt]);
            }
    }

    // ---- normalize and store fp32 output [B][T][64]
    float* orow = out + ((size_t)b * TT + t0) * HEAD;
    float inv[8];
#pragma unroll
    for (int r = 0; r < 8; r++) inv[r] = 1.0f / lrun[r];
#pragma unroll
    for (int nt = 0; nt < 4; nt++)
#pragma unroll
        for (int r = 0; r < 8; r++)
            orow[(size_t)(r + 8 * h) * HEAD + 16 * nt + n] = o[nt][r] * inv[r];
}

// ---------------------------------------------------------------------------
extern "C" void kernel_launch(void* const* d_in, const int* in_sizes, int n_in,
                              void* d_out, int out_size, void* d_ws, size_t ws_size,
                              hipStream_t stream) {
    const float* x  = (const float*)d_in[0];
    const float* Wk = (const float*)d_in[1];
    const float* bk = (const float*)d_in[2];
    const float* Wq = (const float*)d_in[3];
    const float* bq = (const float*)d_in[4];
    const float* Wv = (const float*)d_in[5];
    const float* bv = (const float*)d_in[6];
    float* out = (float*)d_out;

    __bf16* kb = (__bf16*)d_ws;                       // [MTOT][64]
    __bf16* qb = kb + (size_t)MTOT * HEAD;            // [MTOT][64]
    __bf16* vt = qb + (size_t)MTOT * HEAD;            // [B][64][T]
    __bf16* Wt = vt + (size_t)MTOT * HEAD;            // [3][64][1024]

    prep_weights<<<(3 * HEAD * DIN + 255) / 256, 256, 0, stream>>>(Wk, Wq, Wv, Wt);
    proj_kernel<<<dim3(MTOT / 128, 3), 256, 0, stream>>>(x, Wt, bk, bq, bv, kb, qb, vt);
    attn_kernel<<<dim3(TT / 128, BB), 256, 0, stream>>>(kb, qb, vt, out);
}
